// RelationalGraphModel_43585328119849
// MI455X (gfx1250) — matrix operand order repaired and checked by reading, not compiled, not run on hardware
//
#include <hip/hip_runtime.h>
#include <hip/hip_bf16.h>
#include <stdint.h>

// ---------------------------------------------------------------------------
// MI455X (gfx1250) implementation of the relational-graph forward pass.
//
// Roofline: ~75.5 GFLOP, >= ~1 GB unavoidable DRAM traffic on A (fp32 input).
// bf16 WMMA (v_wmma_f32_16x16x32_bf16, fp32 accumulate) makes compute ~30us
// vs ~43us of HBM time => memory-bound, the right spot. fp32 WMMA (16x16x4)
// would be ~8x slower on compute. Weights are pre-converted to bf16 and
// transposed to [N][K] so each lane's B fragment is two contiguous 8-element
// K-runs (lane=N%16, K=(lane>>4)*8+0..7 and 16+(lane>>4)*8+0..7, AMD's table order).
// ---------------------------------------------------------------------------

typedef __bf16         bf16x16 __attribute__((ext_vector_type(16)));
typedef float          f32x8   __attribute__((ext_vector_type(8)));
typedef float          f32x4   __attribute__((ext_vector_type(4)));
typedef unsigned short u16x4   __attribute__((ext_vector_type(4)));
typedef unsigned short u16x8   __attribute__((ext_vector_type(8)));
typedef unsigned short u16x16  __attribute__((ext_vector_type(16)));

#define REPORT_SIZE 8000
#define CODE_SIZE   20000
#define FEAT        256
#define BATCH       4096

__device__ __forceinline__ unsigned short f2bf(float f) {
  // round-to-nearest-even fp32 -> bf16
  unsigned u = __builtin_bit_cast(unsigned, f);
  unsigned r = u + 0x7FFFu + ((u >> 16) & 1u);
  return (unsigned short)(r >> 16);
}

__device__ __forceinline__ u16x16 cat16(u16x8 a, u16x8 b) {
  return __builtin_shufflevector(a, b, 0,1,2,3,4,5,6,7,8,9,10,11,12,13,14,15);
}

__device__ __forceinline__ f32x8 wmma_bf16(u16x16 a, u16x16 b, f32x8 c) {
  return __builtin_amdgcn_wmma_f32_16x16x32_bf16(
      /*neg_a=*/false, __builtin_bit_cast(bf16x16, a),
      /*neg_b=*/false, __builtin_bit_cast(bf16x16, b),
      /*c_mod=*/(short)0, c, /*reuse_a=*/false, /*reuse_b=*/false);
}

// 1/max(d, 1e-10) via v_rcp_f32 (1 ulp; bf16 inputs dominate the error budget)
__device__ __forceinline__ float inv_deg(float d) {
  return __builtin_amdgcn_rcpf(fmaxf(d, 1e-10f));
}

// ---------------------------------------------------------------------------
// Weight prep: src [K][256] fp32 -> dst [256][K] bf16 (K-contiguous rows).
// ---------------------------------------------------------------------------
__global__ void k_convert_t(const float* __restrict__ src,
                            unsigned short* __restrict__ dst, int K) {
  long idx = (long)blockIdx.x * blockDim.x + threadIdx.x;
  if (idx >= (long)K * FEAT) return;
  int  n = (int)(idx & (FEAT - 1));
  long k = idx >> 8;
  dst[(long)n * K + k] = f2bf(src[idx]);
}

// ---------------------------------------------------------------------------
// R_r = relu((A[rb] @ C_weight) / deg) : gathered-row GEMM, M-tile=16 per
// block, 16 waves, each wave owns one 16-wide N-tile. K staged 128 at a time
// into LDS as bf16 (padded stride 136 breaks bank conflicts on the per-lane
// fragment reads: row step = 272B -> bank step 4 mod 64, distinct for 16 M).
// ---------------------------------------------------------------------------
__global__ __launch_bounds__(512, 1)
void k_row_aggregate(const float* __restrict__ A, const int* __restrict__ rb,
                     const unsigned short* __restrict__ Wct,
                     float* __restrict__ Rr) {
  __shared__ __align__(16) unsigned short sA[16 * 136];
  __shared__ float sDeg[16];
  const int tid  = threadIdx.x;
  const int wave = tid >> 5, lane = tid & 31;
  const int mtile = blockIdx.x;
  const int mld = tid >> 5;   // staging row 0..15
  const int kq  = tid & 31;   // staging k-group (4 floats each)
  if (tid < 16) sDeg[tid] = 0.f;

  const float* __restrict__ Ap = A + (long)rb[mtile * 16 + mld] * CODE_SIZE;

  const int h  = lane >> 4;          // lane-half selects K sub-run
  const int ml = lane & 15;          // A-frag M / B-frag N (mod 16)
  const int n  = wave * 16 + ml;     // global output feature
  const unsigned short* __restrict__ Wn = Wct + (long)n * CODE_SIZE;

  f32x8 acc = {};
  float degp = 0.f;
  __syncthreads();

  for (int kb = 0; kb < CODE_SIZE; kb += 128) {  // 156 full + one 32-rem chunk
    const int rem = CODE_SIZE - kb;
    const int ck  = rem < 128 ? rem : 128;
    if (kq * 4 < ck) {
      f32x4 v = *(const f32x4*)(Ap + kb + kq * 4);
      degp += v.x + v.y + v.z + v.w;
      u16x4 p = { f2bf(v.x), f2bf(v.y), f2bf(v.z), f2bf(v.w) };
      *(u16x4*)&sA[mld * 136 + kq * 4] = p;
    }
    if (kb + 128 < CODE_SIZE)
      __builtin_prefetch(Ap + kb + 128 + kq * 4, 0, 3);  // global_prefetch_b8
    __syncthreads();

    const int nks = ck >> 5;
#pragma unroll
    for (int ks = 0; ks < 4; ++ks) {
      if (ks >= nks) break;
      // A frag: lane ml holds M=ml; first 8 elems K=h*8+0..7, next 8 K=16+h*8+0..7
      u16x8 a0 = *(const u16x8*)&sA[ml * 136 + ks * 32 + h * 8];
      u16x8 a1 = *(const u16x8*)&sA[ml * 136 + ks * 32 + 16 + h * 8];
      // B frag: lane holds N=ml; first 8 elems K=h*8+0..7, next 8 K=16+h*8+0..7
      const unsigned short* bp = Wn + kb + ks * 32 + h * 8;
      u16x8 b0 = *(const u16x8*)(bp);
      u16x8 b1 = *(const u16x8*)(bp + 16);
      acc = wmma_bf16(cat16(a0, a1), cat16(b0, b1), acc);
    }
    __syncthreads();
  }

  atomicAdd(&sDeg[mld], degp);
  __syncthreads();
  // one v_rcp per M, then pure multiplies in the store loop
  float inv[8];
#pragma unroll
  for (int v = 0; v < 8; ++v) inv[v] = inv_deg(sDeg[v + 8 * h]);
#pragma unroll
  for (int v = 0; v < 8; ++v) {
    int M = v + 8 * h;                       // C/D layout: M = vgpr + 8*(lane>>4)
    Rr[(long)(mtile * 16 + M) * FEAT + n] = fmaxf(acc[v] * inv[v], 0.f);
  }
}

// ---------------------------------------------------------------------------
// C_i, C_j = relu((A[:,cols].T @ R_weight)/deg), fused: one pass over A's
// rows serves both column gathers (same L2 window: 64 rows = 5 MB, all 256
// blocks sweep K in lock-step order), one B fragment feeds two WMMAs.
// K (reports) staged 64 at a time; 8000/64 = 125 exact chunks.
// ---------------------------------------------------------------------------
__global__ __launch_bounds__(512, 1)
void k_col_aggregate(const float* __restrict__ A, const int* __restrict__ ci,
                     const int* __restrict__ cj,
                     const unsigned short* __restrict__ Wrt,
                     float* __restrict__ Ci, float* __restrict__ Cj) {
  __shared__ __align__(16) unsigned short sAi[16 * 72];
  __shared__ __align__(16) unsigned short sAj[16 * 72];
  __shared__ float sDegI[16], sDegJ[16];
  const int tid  = threadIdx.x;
  const int wave = tid >> 5, lane = tid & 31;
  const int mtile = blockIdx.x;
  const int mld = tid & 15;   // staging batch-row
  const int kg  = tid >> 4;   // 0..31: k = kg*2, kg*2+1
  if (tid < 16) { sDegI[tid] = 0.f; sDegJ[tid] = 0.f; }

  const long colI = ci[mtile * 16 + mld];
  const long colJ = cj[mtile * 16 + mld];

  const int h  = lane >> 4;
  const int ml = lane & 15;
  const int n  = wave * 16 + ml;
  const unsigned short* __restrict__ Wn = Wrt + (long)n * REPORT_SIZE;

  f32x8 accI = {}, accJ = {};
  float degpI = 0.f, degpJ = 0.f;
  __syncthreads();

  for (int kb = 0; kb < REPORT_SIZE; kb += 64) {
    const long r0 = (long)(kb + kg * 2) * CODE_SIZE;
    // scattered column gathers (this IS the DRAM-bound part)
    float vi0 = A[r0 + colI];
    float vi1 = A[r0 + CODE_SIZE + colI];
    float vj0 = A[r0 + colJ];
    float vj1 = A[r0 + CODE_SIZE + colJ];
    degpI += vi0 + vi1;
    degpJ += vj0 + vj1;
    *(unsigned int*)&sAi[mld * 72 + kg * 2] =
        (unsigned)f2bf(vi0) | ((unsigned)f2bf(vi1) << 16);
    *(unsigned int*)&sAj[mld * 72 + kg * 2] =
        (unsigned)f2bf(vj0) | ((unsigned)f2bf(vj1) << 16);
    if (kb + 64 < REPORT_SIZE) {
      __builtin_prefetch(A + r0 + (long)64 * CODE_SIZE + colI, 0, 3);
      __builtin_prefetch(A + r0 + (long)64 * CODE_SIZE + colJ, 0, 3);
    }
    __syncthreads();

#pragma unroll
    for (int ks = 0; ks < 2; ++ks) {
      u16x8 ai0 = *(const u16x8*)&sAi[ml * 72 + ks * 32 + h * 8];
      u16x8 ai1 = *(const u16x8*)&sAi[ml * 72 + ks * 32 + 16 + h * 8];
      u16x8 aj0 = *(const u16x8*)&sAj[ml * 72 + ks * 32 + h * 8];
      u16x8 aj1 = *(const u16x8*)&sAj[ml * 72 + ks * 32 + 16 + h * 8];
      const unsigned short* bp = Wn + kb + ks * 32 + h * 8;
      u16x8 b0 = *(const u16x8*)(bp);
      u16x8 b1 = *(const u16x8*)(bp + 16);
      u16x16 bf = cat16(b0, b1);
      accI = wmma_bf16(cat16(ai0, ai1), bf, accI);   // B reused: 2 WMMA / load
      accJ = wmma_bf16(cat16(aj0, aj1), bf, accJ);
    }
    __syncthreads();
  }

  atomicAdd(&sDegI[mld], degpI);
  atomicAdd(&sDegJ[mld], degpJ);
  __syncthreads();
  float invI[8], invJ[8];
#pragma unroll
  for (int v = 0; v < 8; ++v) {
    invI[v] = inv_deg(sDegI[v + 8 * h]);
    invJ[v] = inv_deg(sDegJ[v + 8 * h]);
  }
#pragma unroll
  for (int v = 0; v < 8; ++v) {
    int M = v + 8 * h;
    long o = (long)(mtile * 16 + M) * FEAT + n;
    Ci[o] = fmaxf(accI[v] * invI[v], 0.f);
    Cj[o] = fmaxf(accJ[v] * invJ[v], 0.f);
  }
}

// ---------------------------------------------------------------------------
// pred_i[b] = dot(R_r[b], C_i[b]); pred_j[b] = dot(R_r[b], C_j[b]).
// One wave32 per batch row; 8 features per lane; xor-shuffle reduction.
// ---------------------------------------------------------------------------
__global__ __launch_bounds__(128, 1)
void k_dot(const float* __restrict__ Rr, const float* __restrict__ Ci,
           const float* __restrict__ Cj, float* __restrict__ out) {
  const int wave = threadIdx.x >> 5, lane = threadIdx.x & 31;
  const int b = blockIdx.x * 4 + wave;
  const long base = (long)b * FEAT + lane * 8;
  f32x4 r0 = *(const f32x4*)(Rr + base), r1 = *(const f32x4*)(Rr + base + 4);
  f32x4 i0 = *(const f32x4*)(Ci + base), i1 = *(const f32x4*)(Ci + base + 4);
  f32x4 j0 = *(const f32x4*)(Cj + base), j1 = *(const f32x4*)(Cj + base + 4);
  float si = r0.x*i0.x + r0.y*i0.y + r0.z*i0.z + r0.w*i0.w
           + r1.x*i1.x + r1.y*i1.y + r1.z*i1.z + r1.w*i1.w;
  float sj = r0.x*j0.x + r0.y*j0.y + r0.z*j0.z + r0.w*j0.w
           + r1.x*j1.x + r1.y*j1.y + r1.z*j1.z + r1.w*j1.w;
#pragma unroll
  for (int off = 16; off; off >>= 1) {
    si += __shfl_xor(si, off, 32);
    sj += __shfl_xor(sj, off, 32);
  }
  if (lane == 0) { out[b] = si; out[BATCH + b] = sj; }
}

// ---------------------------------------------------------------------------
// Workspace layout (bytes, 256-aligned):
//   [0)          Wct  bf16 [256][20000]  10,240,000
//   [10,240,000) Wrt  bf16 [256][8000]    4,096,000
//   [14,336,000) Rr   f32  [4096][256]    4,194,304
//   [18,530,304) Ci   f32  [4096][256]    4,194,304
//   [22,724,608) Cj   f32  [4096][256]    4,194,304   => ~26.9 MB total
// ---------------------------------------------------------------------------
extern "C" void kernel_launch(void* const* d_in, const int* in_sizes, int n_in,
                              void* d_out, int out_size, void* d_ws,
                              size_t ws_size, hipStream_t stream) {
  const float* A   = (const float*)d_in[0];
  const int*   rb  = (const int*)d_in[1];
  const int*   cib = (const int*)d_in[2];
  const int*   cjb = (const int*)d_in[3];
  const float* Rw  = (const float*)d_in[4];
  const float* Cw  = (const float*)d_in[5];

  char* ws = (char*)d_ws;
  unsigned short* Wct = (unsigned short*)(ws);
  unsigned short* Wrt = (unsigned short*)(ws + 10240000);
  float* Rr = (float*)(ws + 14336000);
  float* Ci = (float*)(ws + 18530304);
  float* Cj = (float*)(ws + 22724608);
  float* out = (float*)d_out;

  k_convert_t<<<(CODE_SIZE * FEAT + 255) / 256, 256, 0, stream>>>(Cw, Wct, CODE_SIZE);
  k_convert_t<<<(REPORT_SIZE * FEAT + 255) / 256, 256, 0, stream>>>(Rw, Wrt, REPORT_SIZE);
  k_row_aggregate<<<BATCH / 16, 512, 0, stream>>>(A, rb, Wct, Rr);
  k_col_aggregate<<<BATCH / 16, 512, 0, stream>>>(A, cib, cjb, Wrt, Ci, Cj);
  k_dot<<<BATCH / 4, 128, 0, stream>>>(Rr, Ci, Cj, out);
}
